// Denoiser_6408091205962
// MI455X (gfx1250) — matrix operand order repaired and checked by reading, not compiled, not run on hardware
//
#include <hip/hip_runtime.h>
#include <hip/hip_bf16.h>

// ---------------------------------------------------------------------------
// CDNA5 (gfx1250) fused GNN-denoiser forward.
// All 128-wide GEMMs run on v_wmma_f32_16x16x32_bf16 (bf16 A/B, f32 accum).
// One wave32 per 16-edge (or 16-node) tile; activations staged in LDS.
// ---------------------------------------------------------------------------

typedef __bf16 bf16;
typedef __attribute__((ext_vector_type(16))) __bf16 v16bf;
typedef __attribute__((ext_vector_type(8)))  float  v8f;

#define NN 50000
#define NE 400000

union ABFrag { v16bf v; uint4 u[2]; };

__device__ __forceinline__ float silu_f(float x) {
    return x / (1.0f + __expf(-x));
}

// A fragment (16x32 bf16) from an LDS row-major tile, per ISA 16-bit A layout:
// lanes 0-15 m=lane, VGPR0..3 -> K 0..7, VGPR4..7 -> K 16..23; lanes 16-31 +8.
__device__ __forceinline__ void load_afrag(const bf16* A, int pitch, int kt,
                                           int lane, ABFrag& f) {
    const int m = lane & 15;
    const int gh = (lane >> 4) << 3;           // 0 or 8 (halves)
    const bf16* p = A + m * pitch + kt * 32 + gh;
    f.u[0] = *(const uint4*)(p);               // K offsets gh..gh+7
    f.u[1] = *(const uint4*)(p + 16);          // K offsets 16+gh..16+gh+7
}

// Y[16x128] = A[16xK] * W^T ; W packed in B-fragment order (see pack_weights).
template<int NKT>
__device__ __forceinline__ void gemm_tile(const bf16* A, int pitch,
                                          const uint4* __restrict__ W,
                                          int lane, v8f acc[8]) {
    const v8f z = {0.f, 0.f, 0.f, 0.f, 0.f, 0.f, 0.f, 0.f};
#pragma unroll
    for (int nt = 0; nt < 8; ++nt) acc[nt] = z;
    for (int kt = 0; kt < NKT; ++kt) {
        ABFrag a;
        load_afrag(A, pitch, kt, lane, a);
#pragma unroll
        for (int nt = 0; nt < 8; ++nt) {
            ABFrag b;
            const uint4* bp = W + ((size_t)(nt * NKT + kt) * 32 + lane) * 2;
            b.u[0] = bp[0];
            b.u[1] = bp[1];
            acc[nt] = __builtin_amdgcn_wmma_f32_16x16x32_bf16(
                false, a.v, false, b.v, (short)0, acc[nt], false, false);
        }
    }
}

// ---------------------------------------------------------------------------
// Weight packing: W (out x Kin, f32 row-major) -> bf16 B-fragments.
// B layout (32x16 bf16): lanes 0-15 hold K 0..7,16..23 (elem h -> k = h + (h&8)), lanes 16-31
// hold K 8..15,24..31 (k + 8), as in load_afrag; N = lane&15. Frag order: [nt][kt][lane][h]. Zero-pads K.
// ---------------------------------------------------------------------------
__global__ void pack_weights(const float* __restrict__ W, bf16* __restrict__ dst,
                             int Kin, int NKT) {
    int tid = blockIdx.x * blockDim.x + threadIdx.x;
    int total = 8 * NKT * 512;
    if (tid >= total) return;
    int nt   = tid / (NKT * 512);
    int rem  = tid - nt * (NKT * 512);
    int kt   = rem >> 9;
    int r2   = rem & 511;
    int lane = r2 >> 4;
    int h    = r2 & 15;
    int k    = kt * 32 + (h & 7) + ((h & 8) << 1) + ((lane & 16) >> 1);
    int o    = nt * 16 + (lane & 15);
    float v  = (k < Kin) ? W[o * Kin + k] : 0.0f;
    dst[tid] = (bf16)v;
}

// Gather node tables and convert to bf16 (solvent/solute, 256 cols each).
__global__ void gather_nodes(const float* __restrict__ solv,
                             const float* __restrict__ solu,
                             const int* __restrict__ svidx,
                             const int* __restrict__ suidx,
                             bf16* __restrict__ xsolv, bf16* __restrict__ xsolu) {
    int stride = gridDim.x * blockDim.x;
    for (int i = blockIdx.x * blockDim.x + threadIdx.x; i < NN * 256; i += stride) {
        int n = i >> 8, j = i & 255;
        xsolv[i] = (bf16)solv[(size_t)svidx[n] * 256 + j];
        xsolu[i] = (bf16)solu[(size_t)suidx[n] * 256 + j];
    }
}

// node_attr = node_emb_w @ [xsolv | xsolu] + b  -> bf16 (N x 128)
__global__ __launch_bounds__(32) void node_gemm(const bf16* __restrict__ xsolv,
                                                const bf16* __restrict__ xsolu,
                                                const uint4* __restrict__ wpk,
                                                const float* __restrict__ bias,
                                                bf16* __restrict__ natt) {
    __shared__ __align__(16) bf16 stag[16 * 512];
    __shared__ __align__(16) bf16 otile[16 * 128];
    const int lane = threadIdx.x;
    const int n0 = blockIdx.x * 16;
    const uint4* a4 = (const uint4*)xsolv;   // 32 uint4 per row
    const uint4* b4 = (const uint4*)xsolu;
    uint4* st4 = (uint4*)stag;               // 64 uint4 per row
    for (int i = lane; i < 16 * 64; i += 32) {
        int m = i >> 6, q = i & 63;
        st4[m * 64 + q] = (q < 32) ? a4[(size_t)(n0 + m) * 32 + q]
                                   : b4[(size_t)(n0 + m) * 32 + (q - 32)];
    }
    __syncthreads();
    v8f acc[8];
    gemm_tile<16>(stag, 512, wpk, lane, acc);
    const int nl = lane & 15, g = lane >> 4;
#pragma unroll
    for (int nt = 0; nt < 8; ++nt) {
        int n = nt * 16 + nl;
        float bb = bias[n];
#pragma unroll
        for (int r = 0; r < 8; ++r)
            otile[(8 * g + r) * 128 + n] = (bf16)(acc[nt][r] + bb);
    }
    __syncthreads();
    const uint4* o4 = (const uint4*)otile;
    uint4* d4 = (uint4*)natt;
    for (int i = lane; i < 16 * 16; i += 32) {
        int m = i >> 4, q = i & 15;
        d4[(size_t)(n0 + m) * 16 + q] = o4[m * 16 + q];
    }
}

// BatchNorm stats prepass: recompute h = e2d_w1 @ [xs[row]|xs[col]] + b1,
// accumulate per-column sum / sumsq (f32 atomics, LDS-first).
__global__ __launch_bounds__(32) void bnstats(const bf16* __restrict__ xsolv,
                                              const int* __restrict__ ei,
                                              const uint4* __restrict__ w1,
                                              const float* __restrict__ b1,
                                              float* __restrict__ gsum,
                                              float* __restrict__ gsq) {
    __shared__ __align__(16) bf16 stag[16 * 512];
    __shared__ int ids[16][2];
    __shared__ float csum[128], csq[128];
    const int lane = threadIdx.x;
    const int e0 = blockIdx.x * 16;
    for (int i = lane; i < 128; i += 32) { csum[i] = 0.f; csq[i] = 0.f; }
    if (lane < 16) { ids[lane][0] = ei[e0 + lane]; ids[lane][1] = ei[NE + e0 + lane]; }
    __syncthreads();
    const uint4* xs4 = (const uint4*)xsolv;
    uint4* st4 = (uint4*)stag;
    for (int i = lane; i < 16 * 64; i += 32) {
        int m = i >> 6, q = i & 63;
        st4[m * 64 + q] = xs4[(size_t)ids[m][q >> 5] * 32 + (q & 31)];
    }
    __syncthreads();
    v8f acc[8];
    gemm_tile<16>(stag, 512, w1, lane, acc);
    const int nl = lane & 15;
#pragma unroll
    for (int nt = 0; nt < 8; ++nt) {
        int n = nt * 16 + nl;
        float bb = b1[n], s = 0.f, q = 0.f;
#pragma unroll
        for (int r = 0; r < 8; ++r) { float v = acc[nt][r] + bb; s += v; q += v * v; }
        atomicAdd(&csum[n], s);
        atomicAdd(&csq[n], q);
    }
    __syncthreads();
    for (int i = lane; i < 128; i += 32) {
        atomicAdd(&gsum[i], csum[i]);
        atomicAdd(&gsq[i], csq[i]);
    }
}

__global__ void bnfin(const float* __restrict__ sum, const float* __restrict__ sq,
                      const float* __restrict__ g, const float* __restrict__ b,
                      float* __restrict__ scale, float* __restrict__ shift) {
    int n = threadIdx.x;
    float mu  = sum[n] / (float)NE;
    float var = sq[n] / (float)NE - mu * mu;
    float rs  = rsqrtf(var + 1e-5f);
    float sc  = g[n] * rs;
    scale[n] = sc;
    shift[n] = b[n] - mu * sc;
}

// ---------------------------------------------------------------------------
// Fused edge kernel: geometry -> e2d (GEMM,BN,silu,GEMM) -> Fourier ->
// coff_mlp x2 -> proj MLP -> edge_attr -> score-net -> scatter-add.
// ---------------------------------------------------------------------------
__global__ __launch_bounds__(32) void edge_fused(
    const int* __restrict__ ei, const float* __restrict__ pos,
    const float* __restrict__ wf,
    const bf16* __restrict__ xsolv, const bf16* __restrict__ natt,
    const uint4* __restrict__ wp_e2d1, const float* __restrict__ b_e2d1,
    const float* __restrict__ bnscale, const float* __restrict__ bnshift,
    const uint4* __restrict__ wp_e2d2, const float* __restrict__ b_e2d2,
    const uint4* __restrict__ wp_coff, const float* __restrict__ b_coff,
    const uint4* __restrict__ wp_p1, const float* __restrict__ b_p1,
    const uint4* __restrict__ wp_p2, const float* __restrict__ b_p2,
    const uint4* __restrict__ wp_sn1, const float* __restrict__ b_sn1,
    const float* __restrict__ w_sn2, const float* __restrict__ b_sn2,
    float* __restrict__ scores) {
    __shared__ __align__(16) bf16 stag[16 * 512];   // big activation stage
    __shared__ __align__(16) bf16 projA[16 * 288];  // [psin,pcos,emb_i,emb_j,pad]
    __shared__ float park[16 * 132];                // attr2d, later a2 (padded pitch)
    __shared__ float g_d[16][3], g_cr[16][3], g_vt[16][3];
    __shared__ float s_ci[16][3], s_cj[16][3];
    __shared__ int s_ids[16][2];

    const int lane = threadIdx.x;
    const int e0 = blockIdx.x * 16;
    const int nl = lane & 15;
    const int g = lane >> 4;
    v8f acc[8];

    // ---- geometry (fp32, lanes 0..15, one edge each) ----
    if (lane < 16) {
        const int e = e0 + lane;
        const int r = ei[e], c = ei[NE + e];
        s_ids[lane][0] = r; s_ids[lane][1] = c;
        float prx = pos[3 * r], pry = pos[3 * r + 1], prz = pos[3 * r + 2];
        float pcx = pos[3 * c], pcy = pos[3 * c + 1], pcz = pos[3 * c + 2];
        float dx = prx - pcx, dy = pry - pcy, dz = prz - pcz;
        float inv = 1.0f / (sqrtf(dx * dx + dy * dy + dz * dz) + 1e-5f);
        dx *= inv; dy *= inv; dz *= inv;
        float cx = pry * pcz - prz * pcy;
        float cy = prz * pcx - prx * pcz;
        float cz = prx * pcy - pry * pcx;
        inv = 1.0f / (sqrtf(cx * cx + cy * cy + cz * cz) + 1e-5f);
        cx *= inv; cy *= inv; cz *= inv;
        float vx = dy * cz - dz * cy;
        float vy = dz * cx - dx * cz;
        float vz = dx * cy - dy * cx;
        float ci0 = dx * prx + dy * pry + dz * prz;
        float ci1 = fabsf(cx * prx + cy * pry + cz * prz);
        float ci2 = vx * prx + vy * pry + vz * prz;
        float cj0 = dx * pcx + dy * pcy + dz * pcz;
        float cj1 = fabsf(cx * pcx + cy * pcy + cz * pcz);
        float cj2 = vx * pcx + vy * pcy + vz * pcz;
        float ni = sqrtf(ci0 * ci0 + ci1 * ci1 + ci2 * ci2);
        float nj = sqrtf(cj0 * cj0 + cj1 * cj1 + cj2 * cj2);
        float pcs = (ci0 * cj0 + ci1 * cj1 + ci2 * cj2) / (ni + 1e-5f) / (nj + 1e-5f);
        float t = 1.0f - pcs * pcs;
        float psn = sqrtf(t > 0.f ? t : 0.f);
        g_d[lane][0] = dx; g_d[lane][1] = dy; g_d[lane][2] = dz;
        g_cr[lane][0] = cx; g_cr[lane][1] = cy; g_cr[lane][2] = cz;
        g_vt[lane][0] = vx; g_vt[lane][1] = vy; g_vt[lane][2] = vz;
        s_ci[lane][0] = ci0; s_ci[lane][1] = ci1; s_ci[lane][2] = ci2;
        s_cj[lane][0] = cj0; s_cj[lane][1] = cj1; s_cj[lane][2] = cj2;
        projA[lane * 288 + 0] = (bf16)psn;
        projA[lane * 288 + 1] = (bf16)pcs;
    }
    for (int i = lane; i < 16 * 30; i += 32) {      // zero K-pad cols 258..287
        int m = i / 30, q2 = i - m * 30;
        projA[m * 288 + 258 + q2] = (bf16)0.0f;
    }
    __syncthreads();

    // ---- e2d: A = [xsolv[row] | xsolv[col]] (16 x 512 bf16) ----
    {
        const uint4* xs4 = (const uint4*)xsolv;
        uint4* st4 = (uint4*)stag;
        for (int i = lane; i < 16 * 64; i += 32) {
            int m = i >> 6, q = i & 63;
            st4[m * 64 + q] = xs4[(size_t)s_ids[m][q >> 5] * 32 + (q & 31)];
        }
    }
    __syncthreads();
    gemm_tile<16>(stag, 512, wp_e2d1, lane, acc);
#pragma unroll
    for (int nt = 0; nt < 8; ++nt) {                // BN + silu -> stag[:,0:128]
        const int n = nt * 16 + nl;
        const float sc = bnscale[n], sh = bnshift[n], bb = b_e2d1[n];
#pragma unroll
        for (int r = 0; r < 8; ++r) {
            float v = (acc[nt][r] + bb) * sc + sh;
            stag[(8 * g + r) * 512 + n] = (bf16)silu_f(v);
        }
    }
    __syncthreads();
    gemm_tile<4>(stag, 512, wp_e2d2, lane, acc);
#pragma unroll
    for (int nt = 0; nt < 8; ++nt) {                // edge_attr_2d -> park (f32)
        const int n = nt * 16 + nl;
        const float bb = b_e2d2[n];
#pragma unroll
        for (int r = 0; r < 8; ++r)
            park[(8 * g + r) * 132 + n] = acc[nt][r] + bb;
    }
    __syncthreads();

    // ---- Fourier(coff_i) -> coff_mlp -> emb_i -> projA[:,2:130] ----
    for (int i = lane; i < 16 * 256; i += 32) {
        int m = i >> 8, t = i & 255;
        int cs = t >> 7, j = t & 127;
        float x = cs ? s_ci[m][2] : s_ci[m][0];
        float ang = 6.283185307179586f * wf[j] * x;
        stag[m * 512 + cs * 256 + j]       = (bf16)__sinf(ang);
        stag[m * 512 + cs * 256 + 128 + j] = (bf16)__cosf(ang);
    }
    __syncthreads();
    gemm_tile<16>(stag, 512, wp_coff, lane, acc);
#pragma unroll
    for (int nt = 0; nt < 8; ++nt) {
        const int n = nt * 16 + nl;
        const float bb = b_coff[n];
#pragma unroll
        for (int r = 0; r < 8; ++r)
            projA[(8 * g + r) * 288 + 2 + n] = (bf16)(acc[nt][r] + bb);
    }
    __syncthreads();

    // ---- Fourier(coff_j) -> coff_mlp -> emb_j -> projA[:,130:258] ----
    for (int i = lane; i < 16 * 256; i += 32) {
        int m = i >> 8, t = i & 255;
        int cs = t >> 7, j = t & 127;
        float x = cs ? s_cj[m][2] : s_cj[m][0];
        float ang = 6.283185307179586f * wf[j] * x;
        stag[m * 512 + cs * 256 + j]       = (bf16)__sinf(ang);
        stag[m * 512 + cs * 256 + 128 + j] = (bf16)__cosf(ang);
    }
    __syncthreads();
    gemm_tile<16>(stag, 512, wp_coff, lane, acc);
#pragma unroll
    for (int nt = 0; nt < 8; ++nt) {
        const int n = nt * 16 + nl;
        const float bb = b_coff[n];
#pragma unroll
        for (int r = 0; r < 8; ++r)
            projA[(8 * g + r) * 288 + 130 + n] = (bf16)(acc[nt][r] + bb);
    }
    __syncthreads();

    // ---- proj MLP: edge_attr_3d; edge_attr = 2d+3d -> stag[:,256:384] ----
    gemm_tile<9>(projA, 288, wp_p1, lane, acc);
#pragma unroll
    for (int nt = 0; nt < 8; ++nt) {
        const int n = nt * 16 + nl;
        const float bb = b_p1[n];
#pragma unroll
        for (int r = 0; r < 8; ++r)
            stag[(8 * g + r) * 512 + n] = (bf16)silu_f(acc[nt][r] + bb);
    }
    __syncthreads();
    gemm_tile<4>(stag, 512, wp_p2, lane, acc);
#pragma unroll
    for (int nt = 0; nt < 8; ++nt) {
        const int n = nt * 16 + nl;
        const float bb = b_p2[n];
#pragma unroll
        for (int r = 0; r < 8; ++r) {
            const int m = 8 * g + r;
            stag[m * 512 + 256 + n] = (bf16)(acc[nt][r] + bb + park[m * 132 + n]);
        }
    }
    __syncthreads();

    // ---- score net: A = [natt[row] | natt[col] | edge_attr] (16 x 384) ----
    {
        const uint4* na4 = (const uint4*)natt;      // 16 uint4 per row
        uint4* st4 = (uint4*)stag;
        for (int i = lane; i < 16 * 32; i += 32) {
            int m = i >> 5, q = i & 31;
            st4[m * 64 + q] = na4[(size_t)s_ids[m][q >> 4] * 16 + (q & 15)];
        }
    }
    __syncthreads();
    gemm_tile<12>(stag, 512, wp_sn1, lane, acc);
#pragma unroll
    for (int nt = 0; nt < 8; ++nt) {                // silu -> park (f32 a2)
        const int n = nt * 16 + nl;
        const float bb = b_sn1[n];
#pragma unroll
        for (int r = 0; r < 8; ++r)
            park[(8 * g + r) * 132 + n] = silu_f(acc[nt][r] + bb);
    }
    __syncthreads();

    // ---- sn_w2 (3x128) head + basis combine + scatter-add ----
    if (lane < 16) {
        const int m = lane;
        float c0 = b_sn2[0], c1 = b_sn2[1], c2 = b_sn2[2];
        for (int j = 0; j < 128; ++j) {
            float a = park[m * 132 + j];
            c0 += a * w_sn2[j];
            c1 += a * w_sn2[128 + j];
            c2 += a * w_sn2[256 + j];
        }
        float tx = c0 * g_d[m][0] + c1 * g_cr[m][0] + c2 * g_vt[m][0];
        float ty = c0 * g_d[m][1] + c1 * g_cr[m][1] + c2 * g_vt[m][1];
        float tz = c0 * g_d[m][2] + c1 * g_cr[m][2] + c2 * g_vt[m][2];
        const int r = s_ids[m][0];
        atomicAdd(&scores[3 * r + 0], tx);
        atomicAdd(&scores[3 * r + 1], ty);
        atomicAdd(&scores[3 * r + 2], tz);
    }
}

__global__ void loss_kernel(const float* __restrict__ scores,
                            const float* __restrict__ tgt, float* __restrict__ out) {
    __shared__ float red[256];
    int stride = gridDim.x * blockDim.x;
    float local = 0.f;
    for (int n = blockIdx.x * blockDim.x + threadIdx.x; n < NN; n += stride) {
        float dx = scores[3 * n + 0] - tgt[3 * n + 0];
        float dy = scores[3 * n + 1] - tgt[3 * n + 1];
        float dz = scores[3 * n + 2] - tgt[3 * n + 2];
        local += dx * dx + dy * dy + dz * dz;
    }
    red[threadIdx.x] = local;
    __syncthreads();
    for (int s = 128; s > 0; s >>= 1) {
        if (threadIdx.x < s) red[threadIdx.x] += red[threadIdx.x + s];
        __syncthreads();
    }
    if (threadIdx.x == 0) atomicAdd(out, red[0] / (float)NN);
}

// ---------------------------------------------------------------------------
extern "C" void kernel_launch(void* const* d_in, const int* in_sizes, int n_in,
                              void* d_out, int out_size, void* d_ws, size_t ws_size,
                              hipStream_t stream) {
    (void)in_sizes; (void)n_in; (void)out_size; (void)ws_size;
    const float* solvent = (const float*)d_in[0];
    const float* solute  = (const float*)d_in[1];
    const float* pos     = (const float*)d_in[2];
    const float* post    = (const float*)d_in[3];
    const float* wf      = (const float*)d_in[4];
    const float* coff_w  = (const float*)d_in[5];
    const float* coff_b  = (const float*)d_in[6];
    const float* node_w  = (const float*)d_in[7];
    const float* node_b  = (const float*)d_in[8];
    const float* e2d_w1  = (const float*)d_in[9];
    const float* e2d_b1  = (const float*)d_in[10];
    const float* bn_g    = (const float*)d_in[11];
    const float* bn_b    = (const float*)d_in[12];
    const float* e2d_w2  = (const float*)d_in[13];
    const float* e2d_b2  = (const float*)d_in[14];
    const float* p_w1    = (const float*)d_in[15];
    const float* p_b1    = (const float*)d_in[16];
    const float* p_w2    = (const float*)d_in[17];
    const float* p_b2    = (const float*)d_in[18];
    const float* sn_w1   = (const float*)d_in[19];
    const float* sn_b1   = (const float*)d_in[20];
    const float* sn_w2   = (const float*)d_in[21];
    const float* sn_b2   = (const float*)d_in[22];
    const int*   ei      = (const int*)d_in[23];
    const int*   svidx   = (const int*)d_in[24];
    const int*   suidx   = (const int*)d_in[25];

    // workspace layout (bytes, all 256-aligned); total ~65.3 MB
    constexpr size_t OFF_XSOLV  = 0;
    constexpr size_t OFF_XSOLU  = OFF_XSOLV + (size_t)NN * 256 * 2;
    constexpr size_t OFF_NATT   = OFF_XSOLU + (size_t)NN * 256 * 2;
    constexpr size_t OFF_W_E2D1 = OFF_NATT + (size_t)NN * 128 * 2;
    constexpr size_t OFF_W_COFF = OFF_W_E2D1 + 8 * 16 * 512 * 2;
    constexpr size_t OFF_W_NODE = OFF_W_COFF + 8 * 16 * 512 * 2;
    constexpr size_t OFF_W_E2D2 = OFF_W_NODE + 8 * 16 * 512 * 2;
    constexpr size_t OFF_W_P1   = OFF_W_E2D2 + 8 * 4 * 512 * 2;
    constexpr size_t OFF_W_P2   = OFF_W_P1 + 8 * 9 * 512 * 2;
    constexpr size_t OFF_W_SN1  = OFF_W_P2 + 8 * 4 * 512 * 2;
    constexpr size_t OFF_BNSUM  = OFF_W_SN1 + 8 * 12 * 512 * 2;
    constexpr size_t OFF_BNSQ   = OFF_BNSUM + 512;
    constexpr size_t OFF_BNSC   = OFF_BNSQ + 512;
    constexpr size_t OFF_BNSH   = OFF_BNSC + 512;
    constexpr size_t OFF_SCORES = OFF_BNSH + 512;

    char* ws = (char*)d_ws;
    bf16* xsolv   = (bf16*)(ws + OFF_XSOLV);
    bf16* xsolu   = (bf16*)(ws + OFF_XSOLU);
    bf16* natt    = (bf16*)(ws + OFF_NATT);
    bf16* wp_e2d1 = (bf16*)(ws + OFF_W_E2D1);
    bf16* wp_coff = (bf16*)(ws + OFF_W_COFF);
    bf16* wp_node = (bf16*)(ws + OFF_W_NODE);
    bf16* wp_e2d2 = (bf16*)(ws + OFF_W_E2D2);
    bf16* wp_p1   = (bf16*)(ws + OFF_W_P1);
    bf16* wp_p2   = (bf16*)(ws + OFF_W_P2);
    bf16* wp_sn1  = (bf16*)(ws + OFF_W_SN1);
    float* bnsum  = (float*)(ws + OFF_BNSUM);
    float* bnsq   = (float*)(ws + OFF_BNSQ);
    float* bnsc   = (float*)(ws + OFF_BNSC);
    float* bnsh   = (float*)(ws + OFF_BNSH);
    float* scores = (float*)(ws + OFF_SCORES);

    hipMemsetAsync(scores, 0, (size_t)NN * 3 * sizeof(float), stream);
    hipMemsetAsync(bnsum, 0, 1024, stream);   // bnsum + bnsq
    hipMemsetAsync(d_out, 0, sizeof(float), stream);

    auto pack = [&](const float* W, bf16* dst, int Kin, int NKT) {
        int total = 8 * NKT * 512;
        pack_weights<<<(total + 255) / 256, 256, 0, stream>>>(W, dst, Kin, NKT);
    };
    pack(e2d_w1, wp_e2d1, 512, 16);
    pack(coff_w, wp_coff, 512, 16);
    pack(node_w, wp_node, 512, 16);
    pack(e2d_w2, wp_e2d2, 128, 4);
    pack(p_w1,   wp_p1,   258, 9);
    pack(p_w2,   wp_p2,   128, 4);
    pack(sn_w1,  wp_sn1,  384, 12);

    gather_nodes<<<4096, 256, 0, stream>>>(solvent, solute, svidx, suidx, xsolv, xsolu);
    node_gemm<<<NN / 16, 32, 0, stream>>>(xsolv, xsolu, (const uint4*)wp_node,
                                          node_b, natt);
    bnstats<<<NE / 16, 32, 0, stream>>>(xsolv, ei, (const uint4*)wp_e2d1,
                                        e2d_b1, bnsum, bnsq);
    bnfin<<<1, 128, 0, stream>>>(bnsum, bnsq, bn_g, bn_b, bnsc, bnsh);
    edge_fused<<<NE / 16, 32, 0, stream>>>(
        ei, pos, wf, xsolv, natt,
        (const uint4*)wp_e2d1, e2d_b1, bnsc, bnsh,
        (const uint4*)wp_e2d2, e2d_b2,
        (const uint4*)wp_coff, coff_b,
        (const uint4*)wp_p1, p_b1,
        (const uint4*)wp_p2, p_b2,
        (const uint4*)wp_sn1, sn_b1,
        sn_w2, sn_b2, scores);
    loss_kernel<<<256, 256, 0, stream>>>(scores, post, (float*)d_out);
}
